// LambdaLayer_64183991271706
// MI455X (gfx1250) — matrix operand order repaired and checked by reading, not compiled, not run on hardware
//
#include <hip/hip_runtime.h>

typedef _Float16 f16;
typedef __attribute__((ext_vector_type(16))) _Float16 v16h;
typedef __attribute__((ext_vector_type(8)))  _Float16 v8h;
typedef __attribute__((ext_vector_type(4)))  _Float16 v4h;
typedef __attribute__((ext_vector_type(8)))  float    v8f;
typedef __attribute__((ext_vector_type(4)))  float    v4f;

#define B_   32
#define D_   512
#define N_   1024
#define K_   16
#define H_   4
#define V_   128
#define F_   208      // 16 (keys) + 128 (values) + 64 (q), concatenated features
#define EPS_ 1e-5f

// ---------------------------------------------------------------------------
// 0) Convert concatenated weight matrix [208 x 512] to f16 (read-once, tiny)
// ---------------------------------------------------------------------------
__global__ __launch_bounds__(256) void ll_prep_w(
    const float* __restrict__ Wq, const float* __restrict__ Wk,
    const float* __restrict__ Wv, f16* __restrict__ Wh) {
  int idx = blockIdx.x * 256 + threadIdx.x;            // 208*512 = 106496
  int j = idx >> 9, d = idx & 511;
  float w;
  if (j < 16)       w = Wk[j * D_ + d];
  else if (j < 144) w = Wv[(j - 16) * D_ + d];
  else              w = Wq[(j - 144) * D_ + d];
  Wh[idx] = (f16)w;
}

// ---------------------------------------------------------------------------
// 1) Fused QKV GEMM:  P[b,n,f] = sum_d x[b,n,d] * W_all[f,d]   (f16 WMMA, f32 acc)
//    Block = 256 thr (8 waves), tile M=128 rows(n) x 208 cols(f), K-steps of 32.
//    LDS double-buffered (stage k+1 while computing k); B-fragments pipelined
//    one tile ahead so WMMAs don't drain DScnt to zero each time.
// ---------------------------------------------------------------------------
__global__ __launch_bounds__(256) void ll_gemm_qkv(
    const float* __restrict__ x, const f16* __restrict__ Wh,
    float* __restrict__ P) {
  __shared__ __align__(16) f16 ldsA[2][128 * 40];   // x tile, row stride 40 (pad)
  __shared__ __align__(16) f16 ldsW[2][208 * 40];   // W tile, row stride 40 (pad)

  const int b    = blockIdx.y;
  const int n0   = blockIdx.x * 128;
  const int tid  = threadIdx.x;
  const int lane = tid & 31;
  const int wave = tid >> 5;
  const float* xb = x + (size_t)b * (N_ * D_);   // pure reshape: x == [B][N][D]

  v8f acc[13];
#pragma unroll
  for (int t = 0; t < 13; ++t)
#pragma unroll
    for (int i = 0; i < 8; ++i) acc[t][i] = 0.0f;

  const int xrow  = tid >> 1;        // 0..127
  const int xpart = tid & 1;         // 0..1  (16 floats each)

  auto stage = [&](int kk, int buf) {
    // x tile [128 x 32] f32 -> f16
    const float* src = xb + (size_t)(n0 + xrow) * D_ + kk + xpart * 16;
    f16* dst = &ldsA[buf][xrow * 40 + xpart * 16];
#pragma unroll
    for (int i = 0; i < 4; ++i) {
      v4f f = *(const v4f*)(src + i * 4);
      v4h h; h.x = (f16)f.x; h.y = (f16)f.y; h.z = (f16)f.z; h.w = (f16)f.w;
      *(v4h*)(dst + i * 4) = h;
    }
    if (kk + 32 < D_) __builtin_prefetch(src + 32, 0, 1);  // global_prefetch_b8
    // W tile [208 x 32] f16 (L2-resident, 213 KB total)
    for (int seg = tid; seg < 416; seg += 256) {
      int j = seg >> 1, part = seg & 1;
      const f16* ws = Wh + j * D_ + kk + part * 16;
      f16* wd = &ldsW[buf][j * 40 + part * 16];
      *(v8h*)(wd)     = *(const v8h*)(ws);
      *(v8h*)(wd + 8) = *(const v8h*)(ws + 8);
    }
  };

  stage(0, 0);

  for (int kk = 0; kk < D_; kk += 32) {
    const int buf = (kk >> 5) & 1;
    __syncthreads();                       // staged(buf) visible; prev compute done
    if (kk + 32 < D_) stage(kk + 32, buf ^ 1);

    // ---- A fragment (16x32): lane<16 -> M=lane, K {0..7,16..23}; lane>=16 -> K {8..15,24..31}
    const int arow = wave * 16 + (lane & 15);
    const int ka   = (lane < 16) ? 0 : 8;
    v16h a;
    {
      v8h a0 = *(const v8h*)&ldsA[buf][arow * 40 + ka];
      v8h a1 = *(const v8h*)&ldsA[buf][arow * 40 + ka + 16];
#pragma unroll
      for (int i = 0; i < 8; ++i) { a[i] = a0[i]; a[i + 8] = a1[i]; }
    }
    // ---- B fragments (32x16): lane=col j, K {0..7,16..23} / {8..15,24..31}; pipelined 1 ahead
    const int kb = (lane < 16) ? 0 : 8;
    const int jc = lane & 15;
    auto loadBF = [&](int t) {
      v16h bf;
      v8h b0 = *(const v8h*)&ldsW[buf][(t * 16 + jc) * 40 + kb];
      v8h b1 = *(const v8h*)&ldsW[buf][(t * 16 + jc) * 40 + kb + 16];
#pragma unroll
      for (int i = 0; i < 8; ++i) { bf[i] = b0[i]; bf[i + 8] = b1[i]; }
      return bf;
    };
    v16h bc = loadBF(0);
#pragma unroll
    for (int t = 0; t < 13; ++t) {
      v16h bn = bc;
      if (t < 12) bn = loadBF(t + 1);      // issue loads before consuming bc
      acc[t] = __builtin_amdgcn_wmma_f32_16x16x32_f16(
          false, a, false, bc, (short)0, acc[t], false, false);
      bc = bn;
    }
  }

  // ---- store D: lane<16 -> rows M=0..7 (vgpr i), lane>=16 -> M=8..15; col = lane&15
  const int colw  = lane & 15;
  const int rbase = n0 + wave * 16 + ((lane < 16) ? 0 : 8);
#pragma unroll
  for (int t = 0; t < 13; ++t)
#pragma unroll
    for (int i = 0; i < 8; ++i)
      P[(size_t)b * (N_ * F_) + (size_t)(rbase + i) * F_ + t * 16 + colw] = acc[t][i];
}

// ---------------------------------------------------------------------------
// 2) Softmax over n per (b,k); output sk_f16 laid out [B][K][N] (A-major for WMMA)
// ---------------------------------------------------------------------------
__global__ __launch_bounds__(256) void ll_softmax(
    const float* __restrict__ P, f16* __restrict__ sk16) {
  const int b = blockIdx.x >> 4, k = blockIdx.x & 15, tid = threadIdx.x;
  __shared__ float red[256];
  const float* base = P + (size_t)b * (N_ * F_) + k;
  float vals[4], mx = -1e30f;
#pragma unroll
  for (int i = 0; i < 4; ++i) {
    vals[i] = base[(size_t)(tid + i * 256) * F_];
    mx = fmaxf(mx, vals[i]);
  }
  red[tid] = mx; __syncthreads();
  for (int s = 128; s > 0; s >>= 1) {
    if (tid < s) red[tid] = fmaxf(red[tid], red[tid + s]);
    __syncthreads();
  }
  mx = red[0]; __syncthreads();
  float e[4], sum = 0.f;
#pragma unroll
  for (int i = 0; i < 4; ++i) { e[i] = __expf(vals[i] - mx); sum += e[i]; }
  red[tid] = sum; __syncthreads();
  for (int s = 128; s > 0; s >>= 1) {
    if (tid < s) red[tid] += red[tid + s];
    __syncthreads();
  }
  const float inv = 1.0f / red[0];
#pragma unroll
  for (int i = 0; i < 4; ++i)
    sk16[(size_t)b * (K_ * N_) + k * N_ + tid + i * 256] = (f16)(e[i] * inv);
}

// ---------------------------------------------------------------------------
// 3) BN stats for values: per n over (b, v)  -> mean_v[n], rstd_v[n]
// ---------------------------------------------------------------------------
__global__ __launch_bounds__(256) void ll_bnv_stats(
    const float* __restrict__ P, float* __restrict__ mean_v, float* __restrict__ rstd_v) {
  const int n = blockIdx.x, tid = threadIdx.x;
  __shared__ float s1[256], s2[256];
  float sum = 0.f, sq = 0.f;
#pragma unroll
  for (int i = 0; i < 16; ++i) {
    int idx = tid + i * 256;                  // 4096 = 32b * 128v
    int bb = idx >> 7, v = idx & 127;
    float val = P[(size_t)bb * (N_ * F_) + (size_t)n * F_ + 16 + v];
    sum += val; sq += val * val;
  }
  s1[tid] = sum; s2[tid] = sq; __syncthreads();
  for (int s = 128; s > 0; s >>= 1) {
    if (tid < s) { s1[tid] += s1[tid + s]; s2[tid] += s2[tid + s]; }
    __syncthreads();
  }
  if (tid == 0) {
    float m = s1[0] * (1.0f / 4096.0f);
    float var = s2[0] * (1.0f / 4096.0f) - m * m;
    mean_v[n] = m; rstd_v[n] = rsqrtf(var + EPS_);
  }
}

// ---------------------------------------------------------------------------
// 4) Apply BN-v, write values as f16 [B][N][V]
// ---------------------------------------------------------------------------
__global__ __launch_bounds__(256) void ll_bnv_apply(
    const float* __restrict__ P, const float* __restrict__ mean_v,
    const float* __restrict__ rstd_v, const float* __restrict__ gamma_v,
    const float* __restrict__ beta_v, f16* __restrict__ v16) {
  int idx = blockIdx.x * 256 + threadIdx.x;           // B*N*V = 4,194,304
  int v = idx & 127, n = (idx >> 7) & 1023, b = idx >> 17;
  float val = P[(size_t)b * (N_ * F_) + (size_t)n * F_ + 16 + v];
  v16[idx] = (f16)((val - mean_v[n]) * rstd_v[n] * gamma_v[n] + beta_v[n]);
}

// ---------------------------------------------------------------------------
// 5) BN stats for q: per k over (b, n, h) -> mean_q[k], rstd_q[k]
// ---------------------------------------------------------------------------
__global__ __launch_bounds__(1024) void ll_bnq_stats(
    const float* __restrict__ P, float* __restrict__ mean_q, float* __restrict__ rstd_q) {
  const int k = blockIdx.x, tid = threadIdx.x;
  __shared__ float s1[1024], s2[1024];
  float sum = 0.f, sq = 0.f;
  for (int i = 0; i < 128; ++i) {
    int j = tid + i * 1024;                   // 131072 = 32b * 1024n * 4h
    int bb = j >> 12, nh = j & 4095, n = nh >> 2, h = nh & 3;
    float val = P[(size_t)bb * (N_ * F_) + (size_t)n * F_ + 144 + k * 4 + h];
    sum += val; sq += val * val;
  }
  s1[tid] = sum; s2[tid] = sq; __syncthreads();
  for (int s = 512; s > 0; s >>= 1) {
    if (tid < s) { s1[tid] += s1[tid + s]; s2[tid] += s2[tid + s]; }
    __syncthreads();
  }
  if (tid == 0) {
    float m = s1[0] * (1.0f / 131072.0f);
    float var = s2[0] * (1.0f / 131072.0f) - m * m;
    mean_q[k] = m; rstd_q[k] = rsqrtf(var + EPS_);
  }
}

// ---------------------------------------------------------------------------
// 6) content_lambda[b,k,v] = sum_n sk[b,n,k]*values[b,n,v]   (WMMA, K-loop over n)
//    Block = 256 thr (8 waves) per batch; wave t owns v-tile t.
// ---------------------------------------------------------------------------
__global__ __launch_bounds__(256) void ll_content_lambda(
    const f16* __restrict__ sk16, const f16* __restrict__ v16,
    float* __restrict__ cl) {
  const int b = blockIdx.x, tid = threadIdx.x, lane = tid & 31, wave = tid >> 5;
  __shared__ __align__(16) f16 la[16 * 40];    // sk tile  [16k x 32n]
  __shared__ __align__(16) f16 lb[32 * 136];   // val tile [32n x 128v]
  const f16* skb = sk16 + (size_t)b * (K_ * N_);
  const f16* vb  = v16 + (size_t)b * (N_ * V_);
  v8f acc;
#pragma unroll
  for (int i = 0; i < 8; ++i) acc[i] = 0.0f;

  for (int kk = 0; kk < N_; kk += 32) {
    __syncthreads();
    {   // A: 16x32, thread loads 2 f16
      int kr = tid >> 4, c = (tid & 15) * 2;
      la[kr * 40 + c]     = skb[kr * N_ + kk + c];
      la[kr * 40 + c + 1] = skb[kr * N_ + kk + c + 1];
    }
    {   // B: 32x128, thread copies 16 f16 (32B)
      int r = tid >> 3, seg = tid & 7;
      const f16* src = vb + (size_t)(kk + r) * V_ + seg * 16;
      f16* dst = &lb[r * 136 + seg * 16];
      *(v8h*)(dst)     = *(const v8h*)(src);
      *(v8h*)(dst + 8) = *(const v8h*)(src + 8);
    }
    __syncthreads();

    const int arow = lane & 15, ka = (lane < 16) ? 0 : 8;
    v16h a;
    {
      v8h a0 = *(const v8h*)&la[arow * 40 + ka];
      v8h a1 = *(const v8h*)&la[arow * 40 + ka + 16];
#pragma unroll
      for (int i = 0; i < 8; ++i) { a[i] = a0[i]; a[i + 8] = a1[i]; }
    }
    v16h bf;
    const int vcol = wave * 16 + (lane & 15);
    const int r0 = (lane < 16) ? 0 : 8;
#pragma unroll
    for (int i = 0; i < 16; ++i) bf[i] = lb[(r0 + i + (i >= 8 ? 8 : 0)) * 136 + vcol];
    acc = __builtin_amdgcn_wmma_f32_16x16x32_f16(
        false, a, false, bf, (short)0, acc, false, false);
  }
  const int v = wave * 16 + (lane & 15);
  const int kb = (lane < 16) ? 0 : 8;
#pragma unroll
  for (int i = 0; i < 8; ++i)
    cl[(size_t)b * (K_ * V_) + (kb + i) * V_ + v] = acc[i];
}

// ---------------------------------------------------------------------------
// 7) Fused output: out[b,d=h*V+v,n] = sum_k qBN[b,k,n,h]*cl[b,k,v]
//                                   + sum_p qe[b,n,h,p]*values[b,nbr_p(n),v]
//    Conv3d folded into 9-tap qe (qe = sum_k qBN*E[k,p]); no 256MB pos tensor.
//    Block = 256 thr (8 waves); each wave handles one n; lane covers 4 v.
// ---------------------------------------------------------------------------
__global__ __launch_bounds__(256) void ll_out(
    const float* __restrict__ P, const float* __restrict__ mean_q,
    const float* __restrict__ rstd_q, const float* __restrict__ gamma_q,
    const float* __restrict__ beta_q, const float* __restrict__ cl,
    const f16* __restrict__ v16, const float* __restrict__ E,
    float* __restrict__ out) {
  __shared__ float Es[144];          // E[k][ky][kx] = E[k*9+p]
  __shared__ float G[8][64];         // qBN for 8 n's: [n_local][k*4+h]
  __shared__ float QE[8][40];        // qe: [n_local][h*9+p]
  const int tid = threadIdx.x, lane = tid & 31, wave = tid >> 5;
  const int flat0 = blockIdx.x * 8;
  const int b = flat0 >> 10, nbase = flat0 & 1023;

  if (tid < 144) Es[tid] = E[tid];
  for (int idx = tid; idx < 512; idx += 256) {
    int nl = idx >> 6, f = idx & 63, k = f >> 2;
    float raw = P[(size_t)b * (N_ * F_) + (size_t)(nbase + nl) * F_ + 144 + f];
    G[nl][f] = (raw - mean_q[k]) * rstd_q[k] * gamma_q[k] + beta_q[k];
  }
  __syncthreads();
  for (int idx = tid; idx < 288; idx += 256) {
    int nl = idx / 36, r = idx % 36, h = r / 9, p = r % 9;
    float s = 0.f;
#pragma unroll
    for (int k = 0; k < 16; ++k) s += G[nl][k * 4 + h] * Es[k * 9 + p];
    QE[nl][r] = s;
  }
  __syncthreads();

  const int nl = wave, n = nbase + nl;
  const int n1 = n >> 5, n2 = n & 31;
  const int v0 = lane * 4;
  float acc[4][4];
#pragma unroll
  for (int h = 0; h < 4; ++h)
#pragma unroll
    for (int j = 0; j < 4; ++j) acc[h][j] = 0.0f;

  // content term
  const float* clb = cl + (size_t)b * (K_ * V_);
#pragma unroll
  for (int k = 0; k < 16; ++k) {
    v4f c = *(const v4f*)(clb + k * V_ + v0);
#pragma unroll
    for (int h = 0; h < 4; ++h) {
      float g = G[nl][k * 4 + h];
      acc[h][0] += g * c.x; acc[h][1] += g * c.y;
      acc[h][2] += g * c.z; acc[h][3] += g * c.w;
    }
  }
  // position term (9-tap over spatial neighbors, zero-padded)
  const f16* vb = v16 + (size_t)b * (N_ * V_);
#pragma unroll
  for (int p = 0; p < 9; ++p) {
    int dy = p / 3 - 1, dx = p % 3 - 1;
    int yy = n1 + dy, xx = n2 + dx;
    if ((unsigned)yy < 32u && (unsigned)xx < 32u) {
      v4h hv = *(const v4h*)(vb + (size_t)(yy * 32 + xx) * V_ + v0);
      float f0 = (float)hv.x, f1 = (float)hv.y, f2 = (float)hv.z, f3 = (float)hv.w;
#pragma unroll
      for (int h = 0; h < 4; ++h) {
        float q = QE[nl][h * 9 + p];
        acc[h][0] += q * f0; acc[h][1] += q * f1;
        acc[h][2] += q * f2; acc[h][3] += q * f3;
      }
    }
  }
  // store [B][D][N], d = h*128 + v
  const size_t obase = (size_t)b * (D_ * N_) + n;
#pragma unroll
  for (int h = 0; h < 4; ++h)
#pragma unroll
    for (int j = 0; j < 4; ++j)
      out[obase + (size_t)(h * 128 + v0 + j) * N_] = acc[h][j];
}

// ---------------------------------------------------------------------------
extern "C" void kernel_launch(void* const* d_in, const int* in_sizes, int n_in,
                              void* d_out, int out_size, void* d_ws, size_t ws_size,
                              hipStream_t stream) {
  const float* x       = (const float*)d_in[0];
  const float* Wq      = (const float*)d_in[1];
  const float* Wk      = (const float*)d_in[2];
  const float* Wv      = (const float*)d_in[3];
  const float* gamma_v = (const float*)d_in[4];
  const float* beta_v  = (const float*)d_in[5];
  const float* gamma_q = (const float*)d_in[6];
  const float* beta_q  = (const float*)d_in[7];
  const float* E       = (const float*)d_in[8];
  float* out = (float*)d_out;

  char* ws = (char*)d_ws;
  size_t o = 0;
  float* P      = (float*)(ws + o); o += (size_t)B_ * N_ * F_ * 4;   // 27,262,976
  f16*   Wh     = (f16*)  (ws + o); o += (size_t)F_ * D_ * 2;        //    212,992
  f16*   sk16   = (f16*)  (ws + o); o += (size_t)B_ * K_ * N_ * 2;   //  1,048,576
  f16*   v16    = (f16*)  (ws + o); o += (size_t)B_ * N_ * V_ * 2;   //  8,388,608
  float* cl     = (float*)(ws + o); o += (size_t)B_ * K_ * V_ * 4;   //    262,144
  float* mean_v = (float*)(ws + o); o += N_ * 4;
  float* rstd_v = (float*)(ws + o); o += N_ * 4;
  float* mean_q = (float*)(ws + o); o += 256;
  float* rstd_q = (float*)(ws + o); o += 256;   // total ~37.2 MB

  ll_prep_w<<<(F_ * D_) / 256, 256, 0, stream>>>(Wq, Wk, Wv, Wh);
  ll_gemm_qkv<<<dim3(N_ / 128, B_), 256, 0, stream>>>(x, Wh, P);
  ll_softmax<<<B_ * K_, 256, 0, stream>>>(P, sk16);
  ll_bnv_stats<<<N_, 256, 0, stream>>>(P, mean_v, rstd_v);
  ll_bnv_apply<<<(B_ * N_ * V_) / 256, 256, 0, stream>>>(P, mean_v, rstd_v, gamma_v, beta_v, v16);
  ll_bnq_stats<<<K_, 1024, 0, stream>>>(P, mean_q, rstd_q);
  ll_content_lambda<<<B_, 256, 0, stream>>>(sk16, v16, cl);
  ll_out<<<(B_ * N_) / 8, 256, 0, stream>>>(P, mean_q, rstd_q, gamma_q, beta_q, cl, v16, E, out);
}
